// EncoderBlock_1881195676301
// MI455X (gfx1250) — matrix operand order repaired and checked by reading, not compiled, not run on hardware
//
#include <hip/hip_runtime.h>
#include <math.h>
#include <stdint.h>

// ---------------------------------------------------------------------------
// CDNA5 (gfx1250) transformer encoder block.
//  * All large GEMMs: v_wmma_f32_16x16x32_bf16 (wave32, fp32 accumulate).
//  * GEMM tiles staged global->LDS by the Tensor Data Mover
//    (tensor_load_to_lds, TENSORcnt), double-buffered for copy/compute overlap,
//    with D# LDS padding (4 DW after every 16 DW) giving an 80B row pitch that
//    avoids LDS bank conflicts.
//  * Attention (reference has NO softmax; mask is key-only) is rewritten as
//      out = (1/8) * Q @ (K_masked^T V)  + NEG * sum_{mask==0} V
//    eliminating the [B,H,S,S] score tensor.
// ---------------------------------------------------------------------------

typedef __bf16 bf16;
typedef bf16     bf16x8 __attribute__((ext_vector_type(8)));
typedef bf16     v16bf  __attribute__((ext_vector_type(16)));
typedef float    v8f    __attribute__((ext_vector_type(8)));
typedef uint32_t u32x4  __attribute__((ext_vector_type(4)));
typedef uint32_t u32x8  __attribute__((ext_vector_type(8)));

#define D_MODEL 1024
#define NH      16
#define DK      64
#define D_FF    4096
#define BSZ     4
#define SEQ     1024
#define ROWS    (BSZ * SEQ)      // 4096
#define LN_EPS  1e-5f
#define NEGVAL  (-1e9f)

// ---- WMMA fragment loaders (ISA 7.12.2 layouts, wave32) -------------------
// A (16x32 bf16): half h = lane/16, row = lane%16.
//   element e<8 -> K = 8h + e ; e>=8 -> K = 8h + 16 + (e-8)
__device__ __forceinline__ v16bf load_frag_a(const bf16* rowptr, int hf) {
  bf16x8 lo = *(const bf16x8*)(rowptr + 8 * hf);
  bf16x8 hi = *(const bf16x8*)(rowptr + 8 * hf + 16);
  v16bf f;
#pragma unroll
  for (int i = 0; i < 8; ++i) { f[i] = lo[i]; f[i + 8] = hi[i]; }
  return f;
}
// B (32x16 bf16): half h = lane/16; element e<8 -> K = 8h + e ; e>=8 -> K = 8h + 16 + (e-8);
// column N = lane%16.  rowptr points at Bt[row=n][k0].
__device__ __forceinline__ v16bf load_frag_b(const bf16* rowptr, int hf) {
  bf16x8 lo = *(const bf16x8*)(rowptr + 8 * hf);
  bf16x8 hi = *(const bf16x8*)(rowptr + 8 * hf + 16);
  v16bf f;
#pragma unroll
  for (int i = 0; i < 8; ++i) { f[i] = lo[i]; f[i + 8] = hi[i]; }
  return f;
}

__device__ __forceinline__ v8f zero_v8f() {
  v8f z;
#pragma unroll
  for (int i = 0; i < 8; ++i) z[i] = 0.0f;
  return z;
}

// ---- Tensor Data Mover: 2D tile (rows x 32 bf16) global -> LDS ------------
// D# pad config: pad 4 DWORDs after every 16 DWORDs (one 64B row) -> LDS row
// pitch 80B, matching LDT = 40 bf16.  data_size = 1 (2 bytes).
__device__ __forceinline__ void tdm_load_tile(uint32_t lds_off, const bf16* gptr,
                                              uint32_t lda_elems, uint32_t rows) {
  const uint64_t ga = (uint64_t)(uintptr_t)gptr;
  u32x4 g0;
  g0[0] = 1u;                                                  // count=1, user D#
  g0[1] = lds_off;                                             // LDS byte address
  g0[2] = (uint32_t)ga;                                        // global_addr[31:0]
  g0[3] = (uint32_t)((ga >> 32) & 0x01FFFFFFu) | (2u << 30);   // [56:32] | type=2
  u32x8 g1;
  g1[0] = (1u << 16) | (1u << 20) | (3u << 22) | (3u << 25);   // dsize=2B,pad_en,int=16DW,amt=4DW
  g1[1] = (32u << 16);                                         // tensor_dim0 = 32
  g1[2] = (rows << 16);                                        // tensor_dim1 = rows
  g1[3] = (32u << 16);                                         // tile_dim0 = 32
  g1[4] = rows;                                                // tile_dim1 = rows
  g1[5] = lda_elems;                                           // tensor_dim0_stride
  g1[6] = 0u;
  g1[7] = 0u;
  asm volatile("tensor_load_to_lds %0, %1" :: "s"(g0), "s"(g1) : "memory");
}

// ---- fp32 -> bf16 transposed weight convert (tiled through LDS) -----------
__global__ __launch_bounds__(256)
void transpose_to_bf16_kernel(const float* __restrict__ W, bf16* __restrict__ Wt,
                              int R, int C) {
  __shared__ float tile[32][33];
  const int c0 = blockIdx.x * 32, r0 = blockIdx.y * 32;
  const int tx = threadIdx.x & 31, ty = threadIdx.x >> 5;  // 32 x 8
#pragma unroll
  for (int j = 0; j < 4; ++j)
    tile[ty + 8 * j][tx] = W[(size_t)(r0 + ty + 8 * j) * C + c0 + tx];
  __syncthreads();
#pragma unroll
  for (int j = 0; j < 4; ++j)
    Wt[(size_t)(c0 + ty + 8 * j) * R + r0 + tx] = (bf16)tile[tx][ty + 8 * j];
}

// ---- LayerNorm: one wave per 1024-wide row, torch ddof=1, /(std+eps) ------
__global__ __launch_bounds__(256)
void layernorm_kernel(const float* __restrict__ X, const float* __restrict__ alpha,
                      const float* __restrict__ beta, bf16* __restrict__ out) {
  const int wave = threadIdx.x >> 5, lane = threadIdx.x & 31;
  const int row = blockIdx.x * 8 + wave;
  const float* xr = X + (size_t)row * D_MODEL;
  float v[32], s = 0.f, sq = 0.f;
#pragma unroll
  for (int i = 0; i < 32; ++i) {
    v[i] = xr[i * 32 + lane];
    s += v[i]; sq += v[i] * v[i];
  }
#pragma unroll
  for (int m = 16; m >= 1; m >>= 1) {
    s  += __shfl_xor(s,  m, 32);
    sq += __shfl_xor(sq, m, 32);
  }
  const float mean = s * (1.0f / 1024.0f);
  const float var  = fmaxf(sq - 1024.0f * mean * mean, 0.0f) * (1.0f / 1023.0f);
  const float inv  = alpha[0] / (sqrtf(var) + LN_EPS);
  const float bta  = beta[0];
  bf16* orow = out + (size_t)row * D_MODEL;
#pragma unroll
  for (int i = 0; i < 32; ++i)
    orow[i * 32 + lane] = (bf16)((v[i] - mean) * inv + bta);
}

// ---- Main WMMA GEMM:  C[M,N] = A[M,K] @ Bt[N,K]^T -------------------------
// 256 threads = 8 waves; block tile 128x128, BK=32; wave tile 32x64 (2x4 wmma).
// Tiles DMA'd by the TDM (wave 0 issues; double-buffered; TENSORcnt sync).
// MODE: 0 = normal (outF fp32 / outB bf16); 1 = Kt[b,h,e,s] masked; 2 = Vt.
template <int MODE, bool RELU, bool HAS_RESID, bool OUTF, bool OUTB>
__global__ __launch_bounds__(256)
void gemm_bf16_kernel(const bf16* __restrict__ A, int lda,
                      const bf16* __restrict__ Bt, int ldb,
                      const float* __restrict__ bias,
                      const float* __restrict__ resid,
                      float* __restrict__ outF, bf16* __restrict__ outB,
                      int N, int K,
                      const int* __restrict__ mask, bf16* __restrict__ tdst) {
  constexpr int BM = 128, BN = 128, BK = 32, LDT = BK + 8;  // 80B row pitch
  constexpr int STG = (BM + BN) * LDT;                      // bf16 per stage
  __shared__ bf16 smem[2 * STG];                            // double-buffered A|B

  const int tid  = threadIdx.x;
  const int lane = tid & 31;
  const int waveid = __builtin_amdgcn_readfirstlane(tid >> 5);  // uniform -> scalar branch
  const int wm = waveid & 3, wn = waveid >> 2;  // 4 waves along M, 2 along N
  const int hf = lane >> 4, l16 = lane & 15;
  const int rowBase = blockIdx.y * BM, colBase = blockIdx.x * BN;

  v8f acc[2][4];
#pragma unroll
  for (int i = 0; i < 2; ++i)
#pragma unroll
    for (int j = 0; j < 4; ++j) acc[i][j] = zero_v8f();

  const uint32_t smem_base = (uint32_t)(uintptr_t)smem;
  const int nk = K / BK;

  // Stage 0 prefetch (A tile then B tile; TDM keeps same-wave ordering).
  if (waveid == 0) {
    tdm_load_tile(smem_base, A + (size_t)rowBase * lda, (uint32_t)lda, BM);
    tdm_load_tile(smem_base + BM * LDT * 2, Bt + (size_t)colBase * ldb, (uint32_t)ldb, BN);
  }

  for (int i = 0; i < nk; ++i) {
    if (waveid == 0) __builtin_amdgcn_s_wait_tensorcnt(0);  // stage i resident
    __syncthreads();
    if (waveid == 0 && i + 1 < nk) {                        // prefetch stage i+1
      const int k0 = (i + 1) * BK;
      const uint32_t off = smem_base + ((i + 1) & 1) * STG * 2;
      tdm_load_tile(off, A + (size_t)rowBase * lda + k0, (uint32_t)lda, BM);
      tdm_load_tile(off + BM * LDT * 2, Bt + (size_t)colBase * ldb + k0, (uint32_t)ldb, BN);
    }
    const bf16* As = smem + (i & 1) * STG;
    const bf16* Bs = As + BM * LDT;

    v16bf af[2], bfr[4];
#pragma unroll
    for (int mf = 0; mf < 2; ++mf)
      af[mf] = load_frag_a(&As[(wm * 32 + mf * 16 + l16) * LDT], hf);
#pragma unroll
    for (int nf = 0; nf < 4; ++nf)
      bfr[nf] = load_frag_b(&Bs[(wn * 64 + nf * 16 + l16) * LDT], hf);
#pragma unroll
    for (int mf = 0; mf < 2; ++mf)
#pragma unroll
      for (int nf = 0; nf < 4; ++nf)
        acc[mf][nf] = __builtin_amdgcn_wmma_f32_16x16x32_bf16(
            false, af[mf], false, bfr[nf], (short)0, acc[mf][nf], false, false);
    __syncthreads();
  }

  // Epilogue.  D layout: vgpr i -> row = i + 8*(lane/16), col = lane%16.
#pragma unroll
  for (int mf = 0; mf < 2; ++mf)
#pragma unroll
    for (int nf = 0; nf < 4; ++nf)
#pragma unroll
      for (int i = 0; i < 8; ++i) {
        const int gr = rowBase + wm * 32 + mf * 16 + hf * 8 + i;
        const int gc = colBase + wn * 64 + nf * 16 + l16;
        float v = acc[mf][nf][i] + bias[gc];
        if (RELU) v = fmaxf(v, 0.0f);
        if (HAS_RESID) v += resid[(size_t)gr * N + gc];
        if (MODE == 0) {
          if (OUTF) outF[(size_t)gr * N + gc] = v;
          if (OUTB) outB[(size_t)gr * N + gc] = (bf16)v;
        } else {
          const int b = gr >> 10, s = gr & 1023;   // gr = b*SEQ + s
          const int h = gc >> 6,  e = gc & 63;     // gc = h*DK + e
          if (MODE == 1 && mask[b * SEQ + s] == 0) v = 0.0f;
          tdst[((size_t)((b * NH + h) * DK + e) << 10) + s] = (bf16)v;
        }
      }
}

// ---- Gram kernel: G[b,h][d][e] = sum_s Vt[d,s] * Ktm[e,s]  (64x64, K=1024) -
// 8 waves split K; partial WMMA tiles reduced via LDS float atomics (ds_add).
__global__ __launch_bounds__(256)
void gram_kernel(const bf16* __restrict__ Vt, const bf16* __restrict__ Ktm,
                 bf16* __restrict__ G) {
  const int bh = blockIdx.x;
  const bf16* Arow = Vt  + (size_t)bh * DK * SEQ;  // 64 x 1024 (rows = d)
  const bf16* Brow = Ktm + (size_t)bh * DK * SEQ;  // 64 x 1024 (rows = e, pre-masked)
  const int wave = threadIdx.x >> 5, lane = threadIdx.x & 31;
  const int hf = lane >> 4, l16 = lane & 15;

  v8f acc[4][4];
#pragma unroll
  for (int i = 0; i < 4; ++i)
#pragma unroll
    for (int j = 0; j < 4; ++j) acc[i][j] = zero_v8f();

  const int kbeg = wave * 128;
  for (int k0 = kbeg; k0 < kbeg + 128; k0 += 32) {
    v16bf af[4], bfr[4];
#pragma unroll
    for (int mf = 0; mf < 4; ++mf)
      af[mf] = load_frag_a(Arow + (size_t)(mf * 16 + l16) * SEQ + k0, hf);
#pragma unroll
    for (int nf = 0; nf < 4; ++nf)
      bfr[nf] = load_frag_b(Brow + (size_t)(nf * 16 + l16) * SEQ + k0, hf);
#pragma unroll
    for (int mf = 0; mf < 4; ++mf)
#pragma unroll
      for (int nf = 0; nf < 4; ++nf)
        acc[mf][nf] = __builtin_amdgcn_wmma_f32_16x16x32_bf16(
            false, af[mf], false, bfr[nf], (short)0, acc[mf][nf], false, false);
  }

  __shared__ float Gf[DK * DK];
  for (int i = threadIdx.x; i < DK * DK; i += 256) Gf[i] = 0.0f;
  __syncthreads();
#pragma unroll
  for (int mf = 0; mf < 4; ++mf)
#pragma unroll
    for (int nf = 0; nf < 4; ++nf)
#pragma unroll
      for (int i = 0; i < 8; ++i)
        atomicAdd(&Gf[(mf * 16 + hf * 8 + i) * DK + nf * 16 + l16], acc[mf][nf][i]);
  __syncthreads();
  for (int i = threadIdx.x; i < DK * DK; i += 256)
    G[(size_t)bh * DK * DK + i] = (bf16)Gf[i];
}

// ---- cvec[b,h,d] = NEG * sum_{s: mask==0} V[b,s,h,d] ----------------------
__global__ __launch_bounds__(256)
void cvec_kernel(const bf16* __restrict__ Vt, const int* __restrict__ mask,
                 float* __restrict__ CV) {
  const int wave = threadIdx.x >> 5, lane = threadIdx.x & 31;
  const int row = blockIdx.x * 8 + wave;  // row = bh*64 + d  (4096 total)
  const int b = row >> 10;
  const bf16* vr = Vt + (size_t)row * SEQ;
  const int* mr = mask + b * SEQ;
  float s = 0.0f;
#pragma unroll
  for (int i = 0; i < 32; ++i) {
    const int ss = i * 32 + lane;
    if (mr[ss] == 0) s += (float)vr[ss];
  }
#pragma unroll
  for (int m = 16; m >= 1; m >>= 1) s += __shfl_xor(s, m, 32);
  if (lane == 0) CV[row] = NEGVAL * s;
}

// ---- attention apply:  ATTN[b,s,h*64+d] = 0.125*(Q @ G)[s,d] + cvec[d] ----
__global__ __launch_bounds__(256)
void attn_apply_kernel(const bf16* __restrict__ Q, const bf16* __restrict__ G,
                       const float* __restrict__ CV, bf16* __restrict__ ATTN) {
  const int bh = blockIdx.y, b = bh >> 4, h = bh & 15;
  const int wave = threadIdx.x >> 5, lane = threadIdx.x & 31;
  const int hf = lane >> 4, l16 = lane & 15;
  const int srow = blockIdx.x * 128 + wave * 16;
  const bf16* Abase = Q + ((size_t)(b * SEQ + srow)) * D_MODEL + h * DK;
  const bf16* Bt = G + (size_t)bh * DK * DK;   // rows = d, cols = e
  const float* cv = CV + bh * DK;

  v8f acc[4];
#pragma unroll
  for (int j = 0; j < 4; ++j) acc[j] = zero_v8f();

#pragma unroll
  for (int k0 = 0; k0 < DK; k0 += 32) {
    const v16bf af = load_frag_a(Abase + (size_t)l16 * D_MODEL + k0, hf);
#pragma unroll
    for (int nf = 0; nf < 4; ++nf) {
      const v16bf bfr = load_frag_b(Bt + (size_t)(nf * 16 + l16) * DK + k0, hf);
      acc[nf] = __builtin_amdgcn_wmma_f32_16x16x32_bf16(
          false, af, false, bfr, (short)0, acc[nf], false, false);
    }
  }
#pragma unroll
  for (int nf = 0; nf < 4; ++nf)
#pragma unroll
    for (int i = 0; i < 8; ++i) {
      const int d = nf * 16 + l16;
      const int s = srow + hf * 8 + i;
      const float v = 0.125f * acc[nf][i] + cv[d];   // 1/sqrt(64)
      ATTN[((size_t)(b * SEQ + s)) * D_MODEL + h * DK + d] = (bf16)v;
    }
}

// ---------------------------------------------------------------------------
extern "C" void kernel_launch(void* const* d_in, const int* in_sizes, int n_in,
                              void* d_out, int out_size, void* d_ws, size_t ws_size,
                              hipStream_t stream) {
  (void)in_sizes; (void)n_in; (void)out_size; (void)ws_size;
  const float* x   = (const float*)d_in[0];
  const int*   msk = (const int*)  d_in[1];
  const float* wq  = (const float*)d_in[2];  const float* bq = (const float*)d_in[3];
  const float* wk  = (const float*)d_in[4];  const float* bk = (const float*)d_in[5];
  const float* wv  = (const float*)d_in[6];  const float* bv = (const float*)d_in[7];
  const float* wo  = (const float*)d_in[8];  const float* bo = (const float*)d_in[9];
  const float* w1  = (const float*)d_in[10]; const float* b1 = (const float*)d_in[11];
  const float* w2  = (const float*)d_in[12]; const float* b2 = (const float*)d_in[13];
  const float* ln1a = (const float*)d_in[14]; const float* ln1b = (const float*)d_in[15];
  const float* ln2a = (const float*)d_in[16]; const float* ln2b = (const float*)d_in[17];
  float* out = (float*)d_out;

  char* ws = (char*)d_ws;
  const size_t MB = 1ull << 20;
  bf16*  WQt  = (bf16*)(ws + 0 * MB);     // 1024x1024 bf16 (Bt = W^T)
  bf16*  WKt  = (bf16*)(ws + 2 * MB);
  bf16*  WVt  = (bf16*)(ws + 4 * MB);
  bf16*  WOt  = (bf16*)(ws + 6 * MB);
  bf16*  W1t  = (bf16*)(ws + 8 * MB);     // 4096x1024 bf16
  bf16*  W2t  = (bf16*)(ws + 16 * MB);    // 1024x4096 bf16
  bf16*  XN1  = (bf16*)(ws + 24 * MB);    // 4096x1024
  bf16*  Qb   = (bf16*)(ws + 32 * MB);    // 4096x1024
  bf16*  Ktb  = (bf16*)(ws + 40 * MB);    // [b,h,e,s] pre-masked
  bf16*  Vtb  = (bf16*)(ws + 48 * MB);    // [b,h,d,s]
  bf16*  ATTN = (bf16*)(ws + 56 * MB);    // 4096x1024
  bf16*  XN2  = (bf16*)(ws + 64 * MB);    // 4096x1024
  bf16*  F1   = (bf16*)(ws + 72 * MB);    // 4096x4096
  bf16*  Gb   = (bf16*)(ws + 104 * MB);   // 64 x (64x64)
  float* CV   = (float*)(ws + 105 * MB);  // 64 x 64
  float* X2   = (float*)(ws + 106 * MB);  // 4096x1024 fp32 residual stream

  const dim3 blk(256);

  transpose_to_bf16_kernel<<<dim3(32, 32),  blk, 0, stream>>>(wq, WQt, 1024, 1024);
  transpose_to_bf16_kernel<<<dim3(32, 32),  blk, 0, stream>>>(wk, WKt, 1024, 1024);
  transpose_to_bf16_kernel<<<dim3(32, 32),  blk, 0, stream>>>(wv, WVt, 1024, 1024);
  transpose_to_bf16_kernel<<<dim3(32, 32),  blk, 0, stream>>>(wo, WOt, 1024, 1024);
  transpose_to_bf16_kernel<<<dim3(128, 32), blk, 0, stream>>>(w1, W1t, 1024, 4096);
  transpose_to_bf16_kernel<<<dim3(32, 128), blk, 0, stream>>>(w2, W2t, 4096, 1024);

  layernorm_kernel<<<ROWS / 8, blk, 0, stream>>>(x, ln1a, ln1b, XN1);

  // Q, K(->Kt masked), V(->Vt) projections
  gemm_bf16_kernel<0, false, false, false, true><<<dim3(8, 32), blk, 0, stream>>>(
      XN1, 1024, WQt, 1024, bq, nullptr, nullptr, Qb, 1024, 1024, nullptr, nullptr);
  gemm_bf16_kernel<1, false, false, false, false><<<dim3(8, 32), blk, 0, stream>>>(
      XN1, 1024, WKt, 1024, bk, nullptr, nullptr, nullptr, 1024, 1024, msk, Ktb);
  gemm_bf16_kernel<2, false, false, false, false><<<dim3(8, 32), blk, 0, stream>>>(
      XN1, 1024, WVt, 1024, bv, nullptr, nullptr, nullptr, 1024, 1024, nullptr, Vtb);

  // Linear attention: Gram matrix, constant term, apply.
  gram_kernel<<<BSZ * NH, blk, 0, stream>>>(Vtb, Ktb, Gb);
  cvec_kernel<<<(BSZ * NH * DK) / 8, blk, 0, stream>>>(Vtb, msk, CV);
  attn_apply_kernel<<<dim3(SEQ / 128, BSZ * NH), blk, 0, stream>>>(Qb, Gb, CV, ATTN);

  // Output projection + residual -> X2 (fp32)
  gemm_bf16_kernel<0, false, true, true, false><<<dim3(8, 32), blk, 0, stream>>>(
      ATTN, 1024, WOt, 1024, bo, x, X2, nullptr, 1024, 1024, nullptr, nullptr);

  // LN2 + FFN + residual -> out
  layernorm_kernel<<<ROWS / 8, blk, 0, stream>>>(X2, ln2a, ln2b, XN2);
  gemm_bf16_kernel<0, true, false, false, true><<<dim3(32, 32), blk, 0, stream>>>(
      XN2, 1024, W1t, 1024, b1, nullptr, nullptr, F1, 4096, 1024, nullptr, nullptr);
  gemm_bf16_kernel<0, false, true, true, false><<<dim3(8, 32), blk, 0, stream>>>(
      F1, 4096, W2t, 4096, b2, X2, out, nullptr, 1024, 4096, nullptr, nullptr);
}
